// MixtralAttention_33535104647332
// MI455X (gfx1250) — matrix operand order repaired and checked by reading, not compiled, not run on hardware
//
#include <hip/hip_runtime.h>
#include <hip/hip_bf16.h>

typedef __attribute__((ext_vector_type(16))) __bf16 v16bf;
typedef __attribute__((ext_vector_type(8)))  __bf16 v8bf;
typedef __attribute__((ext_vector_type(8)))  float  v8f;

#define HIDDEN 4096
#define NH     32
#define NKV    8
#define HD     128
#define T_TOK  2048
#define QKV_N  6144   // NH*HD + 2*NKV*HD

// ---------------------------------------------------------------------------
// f32 -> bf16 streaming convert
// ---------------------------------------------------------------------------
__global__ void cvt_f32_bf16(const float* __restrict__ in, __bf16* __restrict__ out, size_t n) {
  for (size_t i = (size_t)blockIdx.x * blockDim.x + threadIdx.x; i < n;
       i += (size_t)gridDim.x * blockDim.x)
    out[i] = (__bf16)in[i];
}

// in: R x C f32 row-major  ->  out: C x R bf16 row-major (i.e. B^T layout)
__global__ void transpose_cvt(const float* __restrict__ in, __bf16* __restrict__ out,
                              int R, int C) {
  __shared__ float tile[32][33];
  const int c0 = blockIdx.x * 32;
  const int r0 = blockIdx.y * 32;
  for (int j = threadIdx.y; j < 32; j += 8)
    tile[j][threadIdx.x] = in[(size_t)(r0 + j) * C + c0 + threadIdx.x];
  __syncthreads();
  for (int j = threadIdx.y; j < 32; j += 8)
    out[(size_t)(c0 + j) * R + r0 + threadIdx.x] = (__bf16)tile[threadIdx.x][j];
}

// ---------------------------------------------------------------------------
// C(f32, MxN) = A(bf16, MxK row-major) * Bt(bf16, NxK row-major)^T
// block = 128 threads (4 waves), block tile 128x128, wave tile 64x64
// ---------------------------------------------------------------------------
__global__ __launch_bounds__(128) void gemm_bf16_tn(
    const __bf16* __restrict__ A, const __bf16* __restrict__ Bt,
    float* __restrict__ C, int M, int N, int K) {
  const int lane = threadIdx.x & 31;
  const int w    = threadIdx.x >> 5;
  const int ll   = lane & 15;
  const int lh   = lane >> 4;
  const int m0 = blockIdx.y * 128 + (w >> 1) * 64;
  const int n0 = blockIdx.x * 128 + (w & 1) * 64;

  v8f acc[4][4] = {};

  for (int k0 = 0; k0 < K; k0 += 32) {
    v16bf a[4], b[4];
#pragma unroll
    for (int mi = 0; mi < 4; ++mi) {
      const __bf16* p = A + (size_t)(m0 + mi * 16 + ll) * K + k0 + lh * 16;
      union { v16bf v; v8bf h[2]; } u;
      u.h[0] = *(const v8bf*)p;          // K = base + 0..7
      u.h[1] = *(const v8bf*)(p + 8);    // K = base + 8..15
      a[mi] = u.v;
    }
#pragma unroll
    for (int ni = 0; ni < 4; ++ni) {
      const __bf16* p = Bt + (size_t)(n0 + ni * 16 + ll) * K + k0 + lh * 16;
      b[ni] = *(const v16bf*)p;          // 16 contiguous K per lane
    }
#pragma unroll
    for (int mi = 0; mi < 4; ++mi)
#pragma unroll
      for (int ni = 0; ni < 4; ++ni)
        acc[mi][ni] = __builtin_amdgcn_wmma_f32_16x16x32_bf16(
            false, a[mi], false, b[ni], (short)0, acc[mi][ni], false, false);
  }

#pragma unroll
  for (int mi = 0; mi < 4; ++mi)
#pragma unroll
    for (int ni = 0; ni < 4; ++ni)
#pragma unroll
      for (int v = 0; v < 8; ++v)
        C[(size_t)(m0 + mi * 16 + v + 8 * lh) * N + n0 + ni * 16 + ll] = acc[mi][ni][v];
}

// ---------------------------------------------------------------------------
// Fused per-head RMSNorm + RoPE, f32 qkv -> bf16 q / k / v^T
// grid (T, 48): slot 0..31 = q head, 32..39 = k head, 40..47 = v head
// ---------------------------------------------------------------------------
__global__ __launch_bounds__(128) void fuse_qkv(
    const float* __restrict__ qkv, const int* __restrict__ positions,
    const float* __restrict__ qw, const float* __restrict__ kw,
    __bf16* __restrict__ qB, __bf16* __restrict__ kB, __bf16* __restrict__ vT) {
  const int t = blockIdx.x;
  const int slot = blockIdx.y;
  const int d = threadIdx.x;

  int type, head, col;
  if (slot < 32)      { type = 0; head = slot;      col = head * HD + d; }
  else if (slot < 40) { type = 1; head = slot - 32; col = NH * HD + head * HD + d; }
  else                { type = 2; head = slot - 40; col = NH * HD + NKV * HD + head * HD + d; }

  const float x = qkv[(size_t)t * QKV_N + col];

  if (type == 2) {  // v: no norm / no rope, store transposed (d-major) for WMMA B operand
    vT[(size_t)(head * HD + d) * T_TOK + t] = (__bf16)x;
    return;
  }

  __shared__ float red[4];
  __shared__ float xs[HD];
  float ss = x * x;
#pragma unroll
  for (int m = 1; m < 32; m <<= 1) ss += __shfl_xor(ss, m, 32);
  if ((threadIdx.x & 31) == 0) red[threadIdx.x >> 5] = ss;
  __syncthreads();
  const float tot = red[0] + red[1] + red[2] + red[3];
  const float inv = rsqrtf(tot * (1.0f / HD) + 1e-6f);
  const float xn = x * inv * ((type == 0) ? qw[d] : kw[d]);
  xs[d] = xn;
  __syncthreads();

  const int i = d & 63;
  const float inv_freq = __powf(1.0e6f, -(float)i * (1.0f / 64.0f));
  const float ang = (float)positions[t] * inv_freq;
  float sv, cv;
  __sincosf(ang, &sv, &cv);
  const float out = (d < 64) ? (xn * cv - xs[d + 64] * sv)
                             : (xn * cv + xs[d - 64] * sv);

  if (type == 0) qB[(size_t)t * (NH * HD) + head * HD + d] = (__bf16)out;
  else           kB[(size_t)t * (NKV * HD) + head * HD + d] = (__bf16)out;
}

// ---------------------------------------------------------------------------
// Flash attention: one wave per (16-query tile, head). Causal, GQA 4:1.
// key tile = 32; scores via 8 WMMAs, O accum via 8 WMMAs per key tile.
// ---------------------------------------------------------------------------
__global__ __launch_bounds__(128) void attn_fa(
    const __bf16* __restrict__ qB, const __bf16* __restrict__ kB,
    const __bf16* __restrict__ vT, __bf16* __restrict__ oB) {
  __shared__ __align__(32) __bf16 pshm[4][16 * 32];
  const int lane = threadIdx.x & 31;
  const int w    = threadIdx.x >> 5;
  const int ll   = lane & 15;
  const int lh   = lane >> 4;
  const int qt   = blockIdx.x;
  const int h    = blockIdx.y * 4 + w;
  const int kvh  = h >> 2;
  const int tq   = qt * 16;

  // Q fragments for this 16-token tile (A operand, K = head dim, 4 chunks of 32)
  v16bf qa[4];
  {
    const __bf16* qrow = qB + (size_t)(tq + ll) * (NH * HD) + h * HD;
#pragma unroll
    for (int kb = 0; kb < 4; ++kb) {
      union { v16bf v; v8bf hf[2]; } u;
      const __bf16* p = qrow + kb * 32 + lh * 16;
      u.hf[0] = *(const v8bf*)p;
      u.hf[1] = *(const v8bf*)(p + 8);
      qa[kb] = u.v;
    }
  }

  v8f O[8] = {};
  float m_r[8], s_r[8];
#pragma unroll
  for (int v = 0; v < 8; ++v) { m_r[v] = -1e30f; s_r[v] = 0.0f; }

  const float scale = 0.08838834764831845f;  // 1/sqrt(128)
  const int nkt = (tq + 15) / 32 + 1;        // causal: key tiles up to the diagonal

  for (int kt = 0; kt < nkt; ++kt) {
    v8f S0 = {}, S1 = {};
#pragma unroll
    for (int kb = 0; kb < 4; ++kb) {
      const __bf16* kp = kB + (size_t)(kt * 32 + ll) * (NKV * HD) + kvh * HD + kb * 32 + lh * 16;
      v16bf b0 = *(const v16bf*)kp;
      v16bf b1 = *(const v16bf*)(kp + (size_t)16 * (NKV * HD));
      S0 = __builtin_amdgcn_wmma_f32_16x16x32_bf16(false, qa[kb], false, b0, (short)0, S0, false, false);
      S1 = __builtin_amdgcn_wmma_f32_16x16x32_bf16(false, qa[kb], false, b1, (short)0, S1, false, false);
    }

    float p0[8], p1[8];
#pragma unroll
    for (int v = 0; v < 8; ++v) {
      const int trow = tq + v + 8 * lh;
      float s0 = S0[v] * scale; if (kt * 32 + ll > trow)      s0 = -1e30f;
      float s1 = S1[v] * scale; if (kt * 32 + 16 + ll > trow) s1 = -1e30f;
      float tm = fmaxf(s0, s1);
      tm = fmaxf(tm, __shfl_xor(tm, 1, 32));
      tm = fmaxf(tm, __shfl_xor(tm, 2, 32));
      tm = fmaxf(tm, __shfl_xor(tm, 4, 32));
      tm = fmaxf(tm, __shfl_xor(tm, 8, 32));
      const float mnew  = fmaxf(m_r[v], tm);
      const float alpha = __expf(m_r[v] - mnew);
      m_r[v] = mnew;
      const float e0 = __expf(s0 - mnew);
      const float e1 = __expf(s1 - mnew);
      p0[v] = e0; p1[v] = e1;
      float rs = e0 + e1;
      rs += __shfl_xor(rs, 1, 32);
      rs += __shfl_xor(rs, 2, 32);
      rs += __shfl_xor(rs, 4, 32);
      rs += __shfl_xor(rs, 8, 32);
      s_r[v] = s_r[v] * alpha + rs;
#pragma unroll
      for (int j = 0; j < 8; ++j) O[j][v] *= alpha;
    }

    // P: C-layout f32 -> LDS -> A-layout bf16 (same-wave DS ops are in order)
    __bf16* pw = pshm[w];
#pragma unroll
    for (int v = 0; v < 8; ++v) {
      pw[(v + 8 * lh) * 32 + ll]      = (__bf16)p0[v];
      pw[(v + 8 * lh) * 32 + 16 + ll] = (__bf16)p1[v];
    }
    asm volatile("s_wait_dscnt 0x0" ::: "memory");
    v16bf aP;
    {
      union { v16bf v; v8bf hf[2]; } u;
      const __bf16* p = pw + ll * 32 + lh * 16;
      u.hf[0] = *(const v8bf*)p;
      u.hf[1] = *(const v8bf*)(p + 8);
      aP = u.v;
    }

#pragma unroll
    for (int j = 0; j < 8; ++j) {
      const __bf16* vp = vT + (size_t)(kvh * HD + j * 16 + ll) * T_TOK + kt * 32 + lh * 16;
      v16bf bv = *(const v16bf*)vp;
      O[j] = __builtin_amdgcn_wmma_f32_16x16x32_bf16(false, aP, false, bv, (short)0, O[j], false, false);
    }
  }

#pragma unroll
  for (int j = 0; j < 8; ++j)
#pragma unroll
    for (int v = 0; v < 8; ++v) {
      const int t = tq + v + 8 * lh;
      oB[(size_t)t * (NH * HD) + h * HD + j * 16 + ll] = (__bf16)(O[j][v] / s_r[v]);
    }
}

// ---------------------------------------------------------------------------
// Launch
// ---------------------------------------------------------------------------
extern "C" void kernel_launch(void* const* d_in, const int* in_sizes, int n_in,
                              void* d_out, int out_size, void* d_ws, size_t ws_size,
                              hipStream_t stream) {
  (void)in_sizes; (void)n_in; (void)out_size; (void)ws_size;
  const int*   positions = (const int*)d_in[0];
  const float* hidden    = (const float*)d_in[1];
  const float* Wqkv      = (const float*)d_in[2];
  const float* Wo        = (const float*)d_in[3];
  const float* qw        = (const float*)d_in[4];
  const float* kw        = (const float*)d_in[5];

  char* ws = (char*)d_ws;
  // Workspace layout (bytes), with aliasing of dead regions:
  //   [0,            16777216): hidden bf16  -> later reused as attention-out bf16
  //   [16777216,     67108864): WqkvT bf16   -> later reused as qB/kB/vT
  //   [67108864,    100663296): WoT bf16
  //   [100663296,   150994944): qkv f32
  __bf16* hidB   = (__bf16*)(ws + 0);
  __bf16* attnB  = (__bf16*)(ws + 0);                  // alias (hidB dead after GEMM1)
  __bf16* wqkvT  = (__bf16*)(ws + 16777216);
  __bf16* qBp    = (__bf16*)(ws + 16777216);           // alias (wqkvT dead after GEMM1)
  __bf16* kBp    = (__bf16*)(ws + 33554432);
  __bf16* vTp    = (__bf16*)(ws + 37748736);
  __bf16* woT    = (__bf16*)(ws + 67108864);
  float*  qkvF   = (float*)(ws + 100663296);

  // 1) precision convert / weight transpose (one streaming pass each)
  cvt_f32_bf16<<<4096, 256, 0, stream>>>(hidden, hidB, (size_t)T_TOK * HIDDEN);
  dim3 tb(32, 8);
  transpose_cvt<<<dim3(QKV_N / 32, HIDDEN / 32), tb, 0, stream>>>(Wqkv, wqkvT, HIDDEN, QKV_N);
  transpose_cvt<<<dim3(HIDDEN / 32, HIDDEN / 32), tb, 0, stream>>>(Wo, woT, HIDDEN, HIDDEN);

  // 2) QKV projection: (2048x4096) x (4096x6144)
  gemm_bf16_tn<<<dim3(QKV_N / 128, T_TOK / 128), 128, 0, stream>>>(
      hidB, wqkvT, qkvF, T_TOK, QKV_N, HIDDEN);

  // 3) per-head RMSNorm + RoPE + bf16 pack (q, k, v^T)
  fuse_qkv<<<dim3(T_TOK, 48), 128, 0, stream>>>(qkvF, positions, qw, kw, qBp, kBp, vTp);

  // 4) causal flash attention, GQA 4:1
  attn_fa<<<dim3(T_TOK / 16, NH / 4), 128, 0, stream>>>(qBp, kBp, vTp, attnB);

  // 5) output projection: (2048x4096) x (4096x4096) -> f32 out
  gemm_bf16_tn<<<dim3(HIDDEN / 128, T_TOK / 128), 128, 0, stream>>>(
      attnB, woT, (float*)d_out, T_TOK, HIDDEN, HIDDEN);
}
